// MLA_5050881540802
// MI455X (gfx1250) — matrix operand order repaired and checked by reading, not compiled, not run on hardware
//
#include <hip/hip_runtime.h>
#include <math.h>

typedef __attribute__((ext_vector_type(16))) __bf16 v16bf;
typedef __attribute__((ext_vector_type(8)))  __bf16 v8bf;
typedef __attribute__((ext_vector_type(8)))  float  v8f;
typedef __attribute__((address_space(3)))    char   lds_char;

// Problem constants (from reference): B,S,H,DQ,DV
constexpr int B_  = 2;
constexpr int S_  = 2048;
constexpr int H_  = 16;
constexpr int DQ_ = 192;
constexpr int DV_ = 128;

constexpr int QTILE = 64;   // queries per workgroup (4 waves x 16)
constexpr int KTILE = 32;   // keys per iteration (WMMA K for bf16)
constexpr float SCALE = 0.0721687836487032f; // 1/sqrt(192)

// ---------------------------------------------------------------------------
// gfx1250 async copy helpers (ASYNCcnt-tracked global->LDS DMA, ISA ch.10/15.18)
// ---------------------------------------------------------------------------
__device__ __forceinline__ unsigned lds_off(const void* p) {
    // generic -> LDS address space -> 32-bit LDS byte address
    return (unsigned)(unsigned long long)(lds_char*)p;
}
__device__ __forceinline__ void async_copy_b128(unsigned lds_addr, const void* gaddr) {
    asm volatile("global_load_async_to_lds_b128 %0, %1, off"
                 :: "v"(lds_addr), "v"(gaddr) : "memory");
}
__device__ __forceinline__ void wait_async0() {
    asm volatile("s_wait_asynccnt 0x0" ::: "memory");
}

// ---------------------------------------------------------------------------
// Pre-pass 1: K fp32 [b,s,h,dq] -> bf16 [b,h,s,dq] (coalesced both sides)
// ---------------------------------------------------------------------------
__global__ void convert_k_kernel(const float* __restrict__ k, __bf16* __restrict__ kbf) {
    size_t idx = (size_t)blockIdx.x * 256 + threadIdx.x;   // output-linear index
    int dq = (int)(idx % DQ_);
    size_t t = idx / DQ_;
    int s = (int)(t % S_);
    size_t t2 = t / S_;
    int h = (int)(t2 % H_);
    int b = (int)(t2 / H_);
    kbf[idx] = (__bf16)k[(((size_t)b * S_ + s) * H_ + h) * DQ_ + dq];
}

// ---------------------------------------------------------------------------
// Pre-pass 2: V fp32 [b,s,h,dv] -> bf16 V^T [b,h,dv,s] (LDS tile transpose)
// ---------------------------------------------------------------------------
__global__ void convert_vt_kernel(const float* __restrict__ v, __bf16* __restrict__ vt) {
    __shared__ float tile[32][33];
    const int tx = threadIdx.x, ty = threadIdx.y;          // 32 x 8
    const int dvt = blockIdx.x % (DV_ / 32);
    const int st  = (blockIdx.x / (DV_ / 32)) % (S_ / 32);
    const int bh  = blockIdx.x / ((DV_ / 32) * (S_ / 32));
    const int h = bh % H_, b = bh / H_;
    const int s0 = st * 32, dv0 = dvt * 32;
#pragma unroll
    for (int r = 0; r < 4; ++r) {
        int s = s0 + ty + r * 8;
        tile[ty + r * 8][tx] = v[(((size_t)b * S_ + s) * H_ + h) * DV_ + dv0 + tx];
    }
    __syncthreads();
#pragma unroll
    for (int r = 0; r < 4; ++r) {
        int dv = dv0 + ty + r * 8;
        vt[(((size_t)b * H_ + h) * DV_ + dv) * S_ + s0 + tx] = (__bf16)tile[tx][ty + r * 8];
    }
}

// ---------------------------------------------------------------------------
// Flash-attention forward. USE_PRE=1: stream pre-converted bf16 K / V^T with
// double-buffered async-to-LDS copies. USE_PRE=0: in-kernel fp32->bf16 staging.
// Template split keeps each instantiation's register allocation lean.
// ---------------------------------------------------------------------------
template <int USE_PRE>
__global__ __launch_bounds__(128, 1)
void mla_fwd_kernel(const float* __restrict__ q,
                    const float* __restrict__ k,
                    const float* __restrict__ v,
                    const __bf16* __restrict__ kbf,
                    const __bf16* __restrict__ vt,
                    float* __restrict__ out)
{
    __shared__ __align__(32) __bf16 lds_k[2][KTILE * DQ_];   // 2 x 12 KB [key][dq]
    __shared__ __align__(32) __bf16 lds_v[2][DV_ * KTILE];   // 2 x  8 KB [dv][key]
    __shared__ __align__(32) __bf16 lds_p[4][16 * KTILE];    // 4 KB per-wave P scratch

    const int tid  = threadIdx.x;
    const int lane = tid & 31;
    const int wave = tid >> 5;
    const int n    = lane & 15;   // column index inside a 16-wide tile
    const int grp  = lane >> 4;   // half-wave group

    const int qb  = blockIdx.x % (S_ / QTILE);
    int tmp       = blockIdx.x / (S_ / QTILE);
    const int h   = tmp % H_;
    const int b   = tmp / H_;
    const int q0  = qb * QTILE + wave * 16;

    const float*  kbase  = k + (((size_t)b * S_) * H_ + h) * DQ_;
    const float*  vbase  = v + (((size_t)b * S_) * H_ + h) * DV_;
    const __bf16* kbbase = kbf + ((size_t)b * H_ + h) * (size_t)S_ * DQ_;  // [s][dq]
    const __bf16* vtbase = vt  + ((size_t)b * H_ + h) * (size_t)DV_ * S_;  // [dv][s]

    // ---- Load Q (16 x 192) into bf16 A-layout registers, pre-scaled.
    v16bf q_a[6];
    {
        const float* qrow = q + (((size_t)b * S_ + (q0 + n)) * H_ + h) * DQ_;
#pragma unroll
        for (int d = 0; d < 6; ++d) {
            const float* p0 = qrow + d * 32 + 16 * grp;
            const float* p1 = qrow + d * 32 + 16 * grp + 8;
            float4 a0 = *(const float4*)(p0);
            float4 a1 = *(const float4*)(p0 + 4);
            float4 b0 = *(const float4*)(p1);
            float4 b1 = *(const float4*)(p1 + 4);
            float lo[8] = {a0.x,a0.y,a0.z,a0.w,a1.x,a1.y,a1.z,a1.w};
            float hi[8] = {b0.x,b0.y,b0.z,b0.w,b1.x,b1.y,b1.z,b1.w};
#pragma unroll
            for (int e = 0; e < 8; ++e) {
                q_a[d][e]     = (__bf16)(lo[e] * SCALE);
                q_a[d][8 + e] = (__bf16)(hi[e] * SCALE);
            }
        }
    }

    v8f o_acc[8];
#pragma unroll
    for (int c = 0; c < 8; ++c) o_acc[c] = {};
    float m_i[8], l_i[8];
#pragma unroll
    for (int r = 0; r < 8; ++r) { m_i[r] = -1e30f; l_i[r] = 0.0f; }

    const int nblocks = (qb * QTILE + QTILE) / KTILE;

    // ---- Async prologue: issue tile 0 into buffer 0 (bf16 tiles, raw copy).
    if constexpr (USE_PRE) {
#pragma unroll
        for (int i = 0; i < 6; ++i) {            // K: 768 x b128
            int idx = tid + i * 128;
            int key = idx / 24, c = idx % 24;    // 24 x 16B per 384B key row
            async_copy_b128(lds_off(&lds_k[0][key * DQ_ + c * 8]),
                            kbbase + (size_t)key * DQ_ + c * 8);
        }
#pragma unroll
        for (int i = 0; i < 4; ++i) {            // V^T: 512 x b128
            int idx = tid + i * 128;
            int dv = idx / 4, c = idx % 4;       // 4 x 16B per 64B dv row
            async_copy_b128(lds_off(&lds_v[0][dv * KTILE + c * 8]),
                            vtbase + (size_t)dv * S_ + c * 8);
        }
    }

    for (int jb = 0; jb < nblocks; ++jb) {
        const int j0 = jb * KTILE;
        const int pi = USE_PRE ? (jb & 1) : 0;

        if constexpr (USE_PRE) {
            wait_async0();          // this wave's tile-jb copies have landed
            __syncthreads();        // all waves' copies landed; prev compute done
            if (jb + 1 < nblocks) { // kick tile jb+1 into the other buffer
                const int po = 1 - pi;
                const int j1 = j0 + KTILE;
#pragma unroll
                for (int i = 0; i < 6; ++i) {
                    int idx = tid + i * 128;
                    int key = idx / 24, c = idx % 24;
                    async_copy_b128(lds_off(&lds_k[po][key * DQ_ + c * 8]),
                                    kbbase + (size_t)(j1 + key) * DQ_ + c * 8);
                }
#pragma unroll
                for (int i = 0; i < 4; ++i) {
                    int idx = tid + i * 128;
                    int dv = idx / 4, c = idx % 4;
                    async_copy_b128(lds_off(&lds_v[po][dv * KTILE + c * 8]),
                                    vtbase + (size_t)dv * S_ + j1 + c * 8);
                }
            }
        } else {
            __syncthreads();
            // K tile (32 x 192 fp32) -> LDS bf16 row-major.
#pragma unroll
            for (int i = 0; i < 12; ++i) {
                int idx = tid + i * 128;
                int key = idx / 48, c4 = idx % 48;
                float4 f = *(const float4*)(kbase + (size_t)(j0 + key) * H_ * DQ_ + c4 * 4);
                __bf16* dst = &lds_k[0][key * DQ_ + c4 * 4];
                dst[0] = (__bf16)f.x; dst[1] = (__bf16)f.y;
                dst[2] = (__bf16)f.z; dst[3] = (__bf16)f.w;
            }
            // V tile (32 x 128 fp32) -> LDS bf16 transposed [dv][key].
#pragma unroll
            for (int i = 0; i < 8; ++i) {
                int idx = tid + i * 128;
                int key = idx / 32, c4 = idx % 32;
                float4 f = *(const float4*)(vbase + (size_t)(j0 + key) * H_ * DV_ + c4 * 4);
                lds_v[0][(c4 * 4 + 0) * KTILE + key] = (__bf16)f.x;
                lds_v[0][(c4 * 4 + 1) * KTILE + key] = (__bf16)f.y;
                lds_v[0][(c4 * 4 + 2) * KTILE + key] = (__bf16)f.z;
                lds_v[0][(c4 * 4 + 3) * KTILE + key] = (__bf16)f.w;
            }
        }
        __syncthreads();

        // ---- S = Q K^T : two 16x16 score tiles.
        v8f s[2];
        s[0] = {}; s[1] = {};
#pragma unroll
        for (int t = 0; t < 2; ++t) {
#pragma unroll
            for (int d = 0; d < 6; ++d) {
                v16bf kb = *(const v16bf*)&lds_k[pi][(t * 16 + n) * DQ_ + d * 32 + grp * 16];
                s[t] = __builtin_amdgcn_wmma_f32_16x16x32_bf16(
                    false, q_a[d], false, kb, (short)0, s[t], false, false);
            }
        }

        // ---- Causal mask + online softmax (row = q0 + r + 8*grp, col = j0 + t*16 + n).
        float mnew[8];
#pragma unroll
        for (int r = 0; r < 8; ++r) {
            const int qg = q0 + r + 8 * grp;
            float s0 = (j0 + n      > qg) ? -1e30f : s[0][r];
            float s1 = (j0 + 16 + n > qg) ? -1e30f : s[1][r];
            s[0][r] = s0; s[1][r] = s1;
            float rm = fmaxf(s0, s1);
            rm = fmaxf(rm, __shfl_xor(rm, 1, 32));
            rm = fmaxf(rm, __shfl_xor(rm, 2, 32));
            rm = fmaxf(rm, __shfl_xor(rm, 4, 32));
            rm = fmaxf(rm, __shfl_xor(rm, 8, 32));
            mnew[r] = fmaxf(m_i[r], rm);
        }

#pragma unroll
        for (int r = 0; r < 8; ++r) {
            const float alpha = __expf(m_i[r] - mnew[r]);
            const float p0 = __expf(s[0][r] - mnew[r]);
            const float p1 = __expf(s[1][r] - mnew[r]);
            float rs = p0 + p1;
            rs += __shfl_xor(rs, 1, 32);
            rs += __shfl_xor(rs, 2, 32);
            rs += __shfl_xor(rs, 4, 32);
            rs += __shfl_xor(rs, 8, 32);
            l_i[r] = l_i[r] * alpha + rs;
            m_i[r] = mnew[r];
#pragma unroll
            for (int c = 0; c < 8; ++c) o_acc[c][r] *= alpha;
            lds_p[wave][(r + 8 * grp) * KTILE + n]      = (__bf16)p0;
            lds_p[wave][(r + 8 * grp) * KTILE + 16 + n] = (__bf16)p1;
        }

        // ---- Reload P as bf16 A-matrix (16 x 32); per-wave LDS, wave-ordered DS ops.
        v16bf pa;
        {
            const __bf16* pr = &lds_p[wave][n * KTILE];
            v8bf plo = *(const v8bf*)(pr + 16 * grp);
            v8bf phi = *(const v8bf*)(pr + 16 * grp + 8);
#pragma unroll
            for (int e = 0; e < 8; ++e) { pa[e] = plo[e]; pa[8 + e] = phi[e]; }
        }

        // ---- O += P V : 8 WMMAs over the 128 dv columns.
#pragma unroll
        for (int c = 0; c < 8; ++c) {
            v16bf vb = *(const v16bf*)&lds_v[pi][(c * 16 + n) * KTILE + grp * 16];
            o_acc[c] = __builtin_amdgcn_wmma_f32_16x16x32_bf16(
                false, pa, false, vb, (short)0, o_acc[c], false, false);
        }
    }

    // ---- Finalize: O /= l, write out [b, q, h, dv].
    float inv_l[8];
#pragma unroll
    for (int r = 0; r < 8; ++r) inv_l[r] = 1.0f / l_i[r];
#pragma unroll
    for (int r = 0; r < 8; ++r) {
        float* orow = out + (((size_t)b * S_ + (q0 + r + 8 * grp)) * H_ + h) * DV_;
#pragma unroll
        for (int c = 0; c < 8; ++c) {
            orow[c * 16 + n] = o_acc[c][r] * inv_l[r];
        }
    }
}

extern "C" void kernel_launch(void* const* d_in, const int* in_sizes, int n_in,
                              void* d_out, int out_size, void* d_ws, size_t ws_size,
                              hipStream_t stream) {
    const float* q = (const float*)d_in[0];
    const float* k = (const float*)d_in[1];
    const float* v = (const float*)d_in[2];
    float* out = (float*)d_out;
    (void)in_sizes; (void)n_in; (void)out_size;

    const size_t k_elems = (size_t)B_ * H_ * S_ * DQ_;
    const size_t v_elems = (size_t)B_ * H_ * DV_ * S_;
    const size_t need    = (k_elems + v_elems) * 2;   // bf16 bytes

    const bool use_pre = (d_ws != nullptr && ws_size >= need);
    __bf16* kbf = (__bf16*)d_ws;
    __bf16* vtp = (__bf16*)((char*)d_ws + k_elems * 2);

    const int grid = B_ * H_ * (S_ / QTILE);   // 1024 workgroups
    if (use_pre) {
        convert_k_kernel<<<(int)(k_elems / 256), 256, 0, stream>>>(k, kbf);
        dim3 tb(32, 8);
        convert_vt_kernel<<<B_ * H_ * (S_ / 32) * (DV_ / 32), tb, 0, stream>>>(v, vtp);
        mla_fwd_kernel<1><<<grid, 128, 0, stream>>>(q, k, v, kbf, vtp, out);
    } else {
        mla_fwd_kernel<0><<<grid, 128, 0, stream>>>(q, k, v, kbf, vtp, out);
    }
}
